// RecurrentNet_82875688943845
// MI455X (gfx1250) — matrix operand order repaired and checked by reading, not compiled, not run on hardware
//
#include <hip/hip_runtime.h>

typedef __attribute__((ext_vector_type(16))) __bf16        v16bf;
typedef __attribute__((ext_vector_type(8)))  float         v8f;
typedef __attribute__((ext_vector_type(4)))  unsigned int  u32x4;

#define B_    64
#define T_    512
#define NIN   256
#define NH    1024
#define NOUT  256
#define BT    (B_ * T_)
#define AL    0.02f

__device__ __forceinline__ unsigned short f2bf(float f) {
    unsigned u = __float_as_uint(f);
    u += 0x7FFFu + ((u >> 16) & 1u);          // round-to-nearest-even
    return (unsigned short)(u >> 16);
}
__device__ __forceinline__ float bf2f(unsigned short h) {
    return __uint_as_float(((unsigned)h) << 16);
}

union FragU { u32x4 u[2]; v16bf v; };          // 32 bytes = 8 VGPRs

// 16-byte async copy global -> LDS (ASYNCcnt-tracked, CDNA5)
__device__ __forceinline__ void async_cp16(const void* gsrc, void* lds_dst) {
    unsigned loff = (unsigned)(unsigned long long)lds_dst;  // low dword = LDS offset
    asm volatile("global_load_async_to_lds_b128 %0, %1, off"
                 :: "v"(loff), "v"(gsrc) : "memory");
}

// ---------------------------------------------------------------- convert
__global__ void convert_f32_bf16(const float* __restrict__ src,
                                 unsigned short* __restrict__ dst, int n) {
    int i = blockIdx.x * blockDim.x + threadIdx.x;
    int stride = gridDim.x * blockDim.x;
    for (; i < n; i += stride) dst[i] = f2bf(src[i]);
}

// ---------------------------------------------------------------- GEMM
// C[M,N] = A[M,K](bf16,rm) @ W[N,K]^T(bf16,rm) + bias[N]
// Block (8 waves / 256 thr) -> 128x64 tile; K-steps of 32 double-buffered
// through LDS via async global->LDS copies. B slab shared by all 8 waves.
template <bool OUT_BF16>
__global__ __launch_bounds__(256) void gemm_bf16(
    const unsigned short* __restrict__ A,  int lda,
    const unsigned short* __restrict__ Bw, int ldb,
    const float* __restrict__ bias,
    void* __restrict__ out, int ldo,
    int K, int numMBlocks)
{
    __shared__ unsigned short Abuf[2][128 * 32];   // 16 KB
    __shared__ unsigned short Bbuf[2][64 * 32];    //  8 KB

    const int tid  = threadIdx.x;
    const int wid  = tid >> 5;
    const int lane = tid & 31;
    const int idx  = lane & 15;
    const int s    = lane >> 4;
    const int mBlock = blockIdx.x % numMBlocks;
    const int nGroup = blockIdx.x / numMBlocks;
    const int mBase  = mBlock * 128;
    const int nBase  = nGroup * 64;

    // stage one 32-wide K-slice of A(128 rows) + B(64 rows): 3 x 16B per thread
    auto issue_tile = [&](int kk, int pbuf) {
        int ci = tid;                              // A: 512 chunks of 16B
#pragma unroll
        for (int r = 0; r < 2; ++r, ci += 256) {
            int row = ci >> 2, q = ci & 3;
            async_cp16(A + (size_t)(mBase + row) * lda + kk + 8 * q,
                       &Abuf[pbuf][row * 32 + 8 * q]);
        }
        int row = tid >> 2, q = tid & 3;           // B: 256 chunks of 16B
        async_cp16(Bw + (size_t)(nBase + row) * ldb + kk + 8 * q,
                   &Bbuf[pbuf][row * 32 + 8 * q]);
    };

    v8f acc[4] = {};
    const int nK = K / 32;
    issue_tile(0, 0);

    for (int kt = 0; kt < nK; ++kt) {
        const int cur = kt & 1;
        if (kt + 1 < nK) {
            issue_tile((kt + 1) * 32, cur ^ 1);
            asm volatile("s_wait_asynccnt 0x3" ::: "memory"); // cur buf landed
        } else {
            asm volatile("s_wait_asynccnt 0x0" ::: "memory");
        }
        __syncthreads();                           // cur buf visible block-wide

        FragU a;                                   // wave's own 16 rows
        const unsigned short* rA = &Abuf[cur][(wid * 16 + idx) * 32];
        a.u[0] = *(const u32x4*)(rA + 8 * s);
        a.u[1] = *(const u32x4*)(rA + 16 + 8 * s);
#pragma unroll
        for (int j = 0; j < 4; ++j) {
            FragU b;                               // shared 64-col B slab
            const unsigned short* rB = &Bbuf[cur][(j * 16 + idx) * 32 + 8 * s];
            b.u[0] = *(const u32x4*)(rB);
            b.u[1] = *(const u32x4*)(rB + 16);
            acc[j] = __builtin_amdgcn_wmma_f32_16x16x32_bf16(
                false, a.v, false, b.v, (short)0, acc[j], false, false);
        }
        __syncthreads();                           // before cur buf is rewritten
    }

#pragma unroll
    for (int j = 0; j < 4; ++j) {
        int col = nBase + j * 16 + idx;
        float bv = bias[col];
#pragma unroll
        for (int v = 0; v < 8; ++v) {
            int m = mBase + wid * 16 + 8 * s + v;
            float val = acc[j][v] + bv;
            if (OUT_BF16)
                ((unsigned short*)out)[(size_t)m * ldo + col] = f2bf(val);
            else
                ((float*)out)[(size_t)m * ldo + col] = val;
        }
    }
}

// ---------------------------------------------------------------- scan
// One workgroup per 16 batch rows; h kept in LDS across all T steps.
// 16 waves; wave w computes h_new columns [64w, 64w+64).
__global__ __launch_bounds__(512) void scan_kernel(
    const unsigned short* __restrict__ x,     // [BT, NH] bf16, row = b*T + t
    const unsigned short* __restrict__ xih,   // [BT, NH] bf16
    const unsigned short* __restrict__ whh,   // [NH, NH] bf16 (L2-resident)
    const float* __restrict__ bhh,            // [NH]
    float* __restrict__ hid_out,              // [BT, NH] f32 (d_out part 1)
    unsigned short* __restrict__ hid_bf)      // [BT, NH] bf16 (for GEMM3)
{
    __shared__ unsigned short h_lds[16][NH];  // 32 KB of 320 KB LDS

    int tid   = threadIdx.x;
    int wid   = tid >> 5;
    int lane  = tid & 31;
    int idx   = lane & 15;
    int s     = lane >> 4;
    int bBase = blockIdx.x * 16;
    int nW    = wid * 64;

    for (int i = tid; i < 16 * NH / 2; i += 512)
        ((unsigned int*)h_lds)[i] = 0u;       // h0 = 0
    __syncthreads();

    for (int t = 0; t < T_; ++t) {
        v8f acc[4] = {};
        for (int kk = 0; kk < NH; kk += 32) {
            FragU a;                                   // A = h tile from LDS
            const unsigned short* rowA = &h_lds[idx][kk];
            a.u[0] = *(const u32x4*)(rowA + 8 * s);
            a.u[1] = *(const u32x4*)(rowA + 16 + 8 * s);
#pragma unroll
            for (int j = 0; j < 4; ++j) {
                FragU b;                               // B = W_hh rows (L2-hot)
                const unsigned short* rowB =
                    whh + (size_t)(nW + j * 16 + idx) * NH + kk + 8 * s;
                b.u[0] = *(const u32x4*)(rowB);
                b.u[1] = *(const u32x4*)(rowB + 16);
                acc[j] = __builtin_amdgcn_wmma_f32_16x16x32_bf16(
                    false, a.v, false, b.v, (short)0, acc[j], false, false);
            }
        }
        __syncthreads();   // all waves done reading h_lds for step t

#pragma unroll
        for (int j = 0; j < 4; ++j) {
            int n = nW + j * 16 + idx;
            float bv = bhh[n];
#pragma unroll
            for (int v = 0; v < 8; ++v) {
                int m = 8 * s + v;
                size_t o = ((size_t)(bBase + m) * T_ + t) * NH + n;
                float cell = acc[j][v] + bf2f(xih[o]) + bv;
                cell = cell > 0.0f ? cell : 0.0f;                 // relu
                float hn = (1.0f - AL) * bf2f(x[o]) + AL * cell;  // leaky blend
                hid_out[o] = hn;
                unsigned short hb = f2bf(hn);
                hid_bf[o] = hb;
                h_lds[m][n] = hb;
            }
        }
        __syncthreads();   // h_lds fully updated before step t+1
    }
}

// ---------------------------------------------------------------- launcher
extern "C" void kernel_launch(void* const* d_in, const int* in_sizes, int n_in,
                              void* d_out, int out_size, void* d_ws, size_t ws_size,
                              hipStream_t stream) {
    const float* input = (const float*)d_in[0];
    const float* Wi    = (const float*)d_in[1];
    const float* bi    = (const float*)d_in[2];
    const float* W_ih  = (const float*)d_in[3];
    const float* b_ih  = (const float*)d_in[4];
    const float* W_hh  = (const float*)d_in[5];
    const float* b_hh  = (const float*)d_in[6];
    const float* Wo    = (const float*)d_in[7];
    const float* bo    = (const float*)d_in[8];

    unsigned short* ws     = (unsigned short*)d_ws;
    unsigned short* in_bf  = ws;                                   //  8,388,608
    unsigned short* wi_bf  = in_bf  + (size_t)BT * NIN;            //    262,144
    unsigned short* wih_bf = wi_bf  + (size_t)NH * NIN;            //  1,048,576
    unsigned short* whh_bf = wih_bf + (size_t)NH * NH;             //  1,048,576
    unsigned short* wo_bf  = whh_bf + (size_t)NH * NH;             //    262,144
    unsigned short* x_bf   = wo_bf  + (size_t)NOUT * NH;           // 33,554,432
    unsigned short* xih_bf = x_bf   + (size_t)BT * NH;             // 33,554,432
    unsigned short* hid_bf = xih_bf + (size_t)BT * NH;             // 33,554,432
    // total ~213 MB of workspace

    float* hid_out = (float*)d_out;                    // hidden_list [B,T,NH]
    float* out_out = (float*)d_out + (size_t)BT * NH;  // output_list [B,T,NOUT]

    convert_f32_bf16<<<1024, 256, 0, stream>>>(input, in_bf,  BT * NIN);
    convert_f32_bf16<<<256,  256, 0, stream>>>(Wi,    wi_bf,  NH * NIN);
    convert_f32_bf16<<<1024, 256, 0, stream>>>(W_ih,  wih_bf, NH * NH);
    convert_f32_bf16<<<1024, 256, 0, stream>>>(W_hh,  whh_bf, NH * NH);
    convert_f32_bf16<<<256,  256, 0, stream>>>(Wo,    wo_bf,  NOUT * NH);

    const int numMBlocks = BT / 128;  // 256

    // GEMM1: x = input @ Wi^T + bi   (M=BT, N=NH, K=NIN)
    gemm_bf16<true><<<numMBlocks * (NH / 64), 256, 0, stream>>>(
        in_bf, NIN, wi_bf, NIN, bi, x_bf, NH, NIN, numMBlocks);

    // GEMM2: xih = x @ W_ih^T + b_ih (M=BT, N=NH, K=NH)
    gemm_bf16<true><<<numMBlocks * (NH / 64), 256, 0, stream>>>(
        x_bf, NH, wih_bf, NH, b_ih, xih_bf, NH, NH, numMBlocks);

    // full recurrent scan in a single launch: 4 workgroups x 16 waves
    scan_kernel<<<B_ / 16, 512, 0, stream>>>(
        x_bf, xih_bf, whh_bf, b_hh, hid_out, hid_bf);

    // GEMM3: out = hidden @ Wo^T + bo (M=BT, N=NOUT, K=NH), f32 output
    gemm_bf16<false><<<numMBlocks * (NOUT / 64), 256, 0, stream>>>(
        hid_bf, NH, wo_bf, NH, bo, out_out, NOUT, NH, numMBlocks);
}
